// cross_attention_39101382263234
// MI455X (gfx1250) — matrix operand order repaired and checked by reading, not compiled, not run on hardware
//
#include <hip/hip_runtime.h>

// Cross-attention block for MI455X (gfx1250, wave32, WMMA).
// GEMMs: v_wmma_f32_16x16x32_bf16 on LDS-staged 128x128 workgroup tiles.
// K staged in 32-deep TRIPLE-buffered LDS tiles filled by CDNA5
// GLOBAL_LOAD_ASYNC_TO_LDS_B128 with two-stage lookahead:
//   stage s issues DMA for stage s+2; "s_wait_asynccnt 4" + barrier proves
//   the next stage's buffer landed (4 async ops/stage, in-order completion)
//   while the newest DMA group stays in flight.
// Softmax / l2norms in fp32.  Workspace: ~583 MB.

#define B_    16
#define DIM_  1024
#define L_    1024

#define LDA_S 40    // 32 + 8 pad  (lane stride 20 dwords -> conflict-free)
#define LDB_S 136   // 128 + 8 pad (lane stride 68 dwords -> conflict-free)

typedef __attribute__((ext_vector_type(16))) __bf16 v16bf;
typedef __attribute__((ext_vector_type(8)))  __bf16 v8bf;
typedef __attribute__((ext_vector_type(8)))  float  v8f;

__device__ __forceinline__ unsigned short f2bf(float f) {
    unsigned int u = __float_as_uint(f);
    u += 0x7FFFu + ((u >> 16) & 1u);         // round-to-nearest-even
    return (unsigned short)(u >> 16);
}

// Low 32 bits of a flat shared-space address = LDS byte offset (ISA 10.2).
__device__ __forceinline__ unsigned lds_off(const void* p) {
    return (unsigned)(unsigned long long)(size_t)p;
}

// Async DMA: copy 32 contiguous bytes global -> LDS (2 x B128, per-lane).
// INST_OFFSET is applied to both the global and the LDS address (ISA 15.18).
__device__ __forceinline__ void async_copy32(unsigned lds_addr, const void* gp) {
    unsigned long long ga = (unsigned long long)(size_t)gp;
    asm volatile("global_load_async_to_lds_b128 %0, %1, off"
                 :: "v"(lds_addr), "v"(ga) : "memory");
    asm volatile("global_load_async_to_lds_b128 %0, %1, off offset:16"
                 :: "v"(lds_addr), "v"(ga) : "memory");
}

__device__ __forceinline__ void wait_async_le4() {
    asm volatile("s_wait_asynccnt 0x4" ::: "memory");
}

// ---------------------------------------------------------------------------
// Workgroup GEMM core: C(128x128,f32) += A(128xK,bf16) * B(Kx128,bf16).
// 256 threads = 8 waves as 2x4 grid of 64x32 wave tiles (acc[4][2]).
// CDNA5 wave32 fragment layouts per ISA 7.12.2:
//   A 16x32: lane m=lane&15 holds row m; K-halves selected by lane&16.
//   B 32x16: lane n=lane&15 holds column n; K split by lane&16 as for A.
//   C 16x16: VGPR r <-> row r + (lane>=16 ? 8 : 0), col = lane&15.
// ---------------------------------------------------------------------------
__device__ __forceinline__ void wg_gemm_128x128(
    const unsigned short* __restrict__ A, int lda,
    const unsigned short* __restrict__ Bm, int ldb,
    int rowWG, int colWG, int K, v8f acc[4][2])
{
    __shared__ unsigned short ldsA[3][128 * LDA_S];   // 30.0 KB
    __shared__ unsigned short ldsB[3][ 32 * LDB_S];   // 25.5 KB

    const int tid   = threadIdx.x;
    const int lane  = tid & 31;
    const int wave  = tid >> 5;
    const int row0w = (wave >> 2) * 64;   // wave M origin within WG tile
    const int col0w = (wave & 3) * 32;    // wave N origin within WG tile

    // staging: each thread DMAs 16 contiguous elements (32 B) of A and of B
    const int aRow = tid >> 1;            // 128 rows x 2 half-rows
    const int aCol = (tid & 1) * 16;
    const int bRow = tid >> 3;            // 32 rows x 8 chunks
    const int bCol = (tid & 7) * 16;

    const unsigned short* gA = A  + (size_t)(rowWG + aRow) * lda + aCol;
    const unsigned short* gB = Bm + (size_t)bRow * ldb + colWG + bCol;

    const unsigned la[3] = { lds_off(&ldsA[0][aRow * LDA_S + aCol]),
                             lds_off(&ldsA[1][aRow * LDA_S + aCol]),
                             lds_off(&ldsA[2][aRow * LDA_S + aCol]) };
    const unsigned lb[3] = { lds_off(&ldsB[0][bRow * LDB_S + bCol]),
                             lds_off(&ldsB[1][bRow * LDB_S + bCol]),
                             lds_off(&ldsB[2][bRow * LDB_S + bCol]) };

    const int nstages = K >> 5;

    // prologue: DMA stages 0 and 1 (4 async ops each)
    async_copy32(la[0], gA);
    async_copy32(lb[0], gB);
    if (nstages > 1) {
        async_copy32(la[1], gA + 32);
        async_copy32(lb[1], gB + (size_t)32 * ldb);
    }
    wait_async_le4();      // stage-0 group retired (in-order completion)
    __syncthreads();

    const int m  = lane & 15;
    const int kb = (lane & 16) ? 8 : 0;

    int bc = 0;            // buffer being computed from
    int bd = 2;            // buffer being DMA-filled (stage s+2)
    for (int s = 0; s < nstages; ++s) {
        if (s + 2 < nstages) {
            // bd was last read at stage s-1; its readers finished before the
            // stage s-1 barrier, so the DMA below cannot race them.
            const int ko = (s + 2) * 32;
            async_copy32(la[bd], gA + ko);
            async_copy32(lb[bd], gB + (size_t)ko * ldb);
        }

        // ---- compute one K=32 slab from LDS buffer bc ----
        v16bf bfv[2];
#pragma unroll
        for (int tj = 0; tj < 2; ++tj) {
            const unsigned short* p = &ldsB[bc][kb * LDB_S + col0w + tj * 16 + m];
#pragma unroll
            for (int s2 = 0; s2 < 8; ++s2) {
                bfv[tj][s2]     = __builtin_bit_cast(__bf16, p[s2 * LDB_S]);
                bfv[tj][s2 + 8] = __builtin_bit_cast(__bf16, p[(s2 + 16) * LDB_S]);
            }
        }
#pragma unroll
        for (int ti = 0; ti < 4; ++ti) {
            const unsigned short* p = &ldsA[bc][(row0w + ti * 16 + m) * LDA_S + kb];
            v8bf lo = *reinterpret_cast<const v8bf*>(p);
            v8bf hi = *reinterpret_cast<const v8bf*>(p + 16);
            v16bf af = __builtin_shufflevector(lo, hi, 0,1,2,3,4,5,6,7,8,9,10,11,12,13,14,15);
            acc[ti][0] = __builtin_amdgcn_wmma_f32_16x16x32_bf16(
                false, af, false, bfv[0], (short)0, acc[ti][0], false, false);
            acc[ti][1] = __builtin_amdgcn_wmma_f32_16x16x32_bf16(
                false, af, false, bfv[1], (short)0, acc[ti][1], false, false);
        }

        // <=4 outstanding: the group filling the NEXT stage's buffer is done;
        // only the group issued this stage may remain in flight.
        wait_async_le4();
        __syncthreads();

        bc = (bc == 2) ? 0 : bc + 1;
        bd = (bd == 2) ? 0 : bd + 1;
    }
}

// ---------------------------------------------------------------------------
// Projection: Y[b] = relu(W (DIMxDIM) * X[b] (DIMxL)) -> bf16
// transposeOut=0: Y[b][e][l] (DIMxL);  transposeOut=1: Y[b][l][e] (LxDIM)
// grid (L/128, DIM/128, B), block 256
// ---------------------------------------------------------------------------
__global__ __launch_bounds__(256) void gemm_proj_kernel(
    const unsigned short* __restrict__ W,
    const unsigned short* __restrict__ X,
    unsigned short* __restrict__ Y,
    int transposeOut)
{
    const int b     = blockIdx.z;
    const int rowWG = blockIdx.y * 128;
    const int colWG = blockIdx.x * 128;

    v8f acc[4][2] = {};
    wg_gemm_128x128(W, DIM_, X + (size_t)b * DIM_ * L_, L_, rowWG, colWG, DIM_, acc);

    const int lane = threadIdx.x & 31;
    const int wave = threadIdx.x >> 5;
    const int n    = lane & 15;
    const int mh   = (lane & 16) ? 8 : 0;
    const int row0 = rowWG + (wave >> 2) * 64;
    const int col0 = colWG + (wave & 3) * 32;
    unsigned short* Yb = Y + (size_t)b * DIM_ * L_;
#pragma unroll
    for (int ti = 0; ti < 4; ++ti)
#pragma unroll
        for (int tj = 0; tj < 2; ++tj) {
            const int colg = col0 + tj * 16 + n;
            const int rowb = row0 + ti * 16 + mh;
#pragma unroll
            for (int r = 0; r < 8; ++r) {
                float v = acc[ti][tj][r];
                v = v > 0.f ? v : 0.f;
                unsigned short h = f2bf(v);
                if (transposeOut) Yb[(size_t)colg * DIM_ + rowb + r] = h;
                else              Yb[(size_t)(rowb + r) * L_ + colg] = h;
            }
        }
}

// ---------------------------------------------------------------------------
// Scores: St[b][k][q] = (1/32) * sum_d Qt[b][q][d] * K[b][d][k]
// Stored k-major so the axis-1 (over q) softmax is a contiguous row op.
// ---------------------------------------------------------------------------
__global__ __launch_bounds__(256) void gemm_qk_kernel(
    const unsigned short* __restrict__ Qt,   // [B][L][DIM]
    const unsigned short* __restrict__ Kx,   // [B][DIM][L]
    float* __restrict__ St)                  // [B][L][L], row=k, col=q
{
    const int b     = blockIdx.z;
    const int rowWG = blockIdx.y * 128;      // q
    const int colWG = blockIdx.x * 128;      // k

    v8f acc[4][2] = {};
    wg_gemm_128x128(Qt + (size_t)b * L_ * DIM_, DIM_,
                    Kx + (size_t)b * DIM_ * L_, L_, rowWG, colWG, DIM_, acc);

    const int lane = threadIdx.x & 31;
    const int wave = threadIdx.x >> 5;
    const int n    = lane & 15;
    const int mh   = (lane & 16) ? 8 : 0;
    const int row0 = rowWG + (wave >> 2) * 64;
    const int col0 = colWG + (wave & 3) * 32;
    float* Sb = St + (size_t)b * L_ * L_;
#pragma unroll
    for (int ti = 0; ti < 4; ++ti)
#pragma unroll
        for (int tj = 0; tj < 2; ++tj) {
            const int colg = col0 + tj * 16 + n;     // k index
            const int rowb = row0 + ti * 16 + mh;    // q base (8 consecutive)
            v8f v = acc[ti][tj] * 0.03125f;          // 1/sqrt(1024)
            *reinterpret_cast<v8f*>(Sb + (size_t)colg * L_ + rowb) = v;
        }
}

// ---------------------------------------------------------------------------
// U[b] = R[b] + Vx[b] (DIMxL) * Pt[b]^T  where Pt is [k][q] row-major,
// i.e. U[d][q] = R[d][q] + sum_k Vx[d][k] * P[q][k]   (= resid + w^T)
// ---------------------------------------------------------------------------
__global__ __launch_bounds__(256) void gemm_pv_kernel(
    const unsigned short* __restrict__ Vx,   // [B][DIM][L]
    const unsigned short* __restrict__ Pt,   // [B][L][L] row=k, col=q (bf16)
    const float* __restrict__ R,             // [B][DIM][L] residual
    float* __restrict__ U)                   // [B][DIM][L]
{
    const int b     = blockIdx.z;
    const int rowWG = blockIdx.y * 128;      // d
    const int colWG = blockIdx.x * 128;      // q

    v8f acc[4][2] = {};
    wg_gemm_128x128(Vx + (size_t)b * DIM_ * L_, L_,
                    Pt + (size_t)b * L_ * L_,   L_, rowWG, colWG, L_, acc);

    const int lane = threadIdx.x & 31;
    const int wave = threadIdx.x >> 5;
    const int n    = lane & 15;
    const int mh   = (lane & 16) ? 8 : 0;
    const int row0 = rowWG + (wave >> 2) * 64;
    const int col0 = colWG + (wave & 3) * 32;
    const size_t off = (size_t)b * DIM_ * L_;
#pragma unroll
    for (int ti = 0; ti < 4; ++ti)
#pragma unroll
        for (int tj = 0; tj < 2; ++tj) {
            const int colg = col0 + tj * 16 + n;
            const int rowb = row0 + ti * 16 + mh;
#pragma unroll
            for (int r = 0; r < 8; ++r) {
                const size_t idx = off + (size_t)(rowb + r) * L_ + colg;
                U[idx] = R[idx] + acc[ti][tj][r];
            }
        }
}

// ---------------------------------------------------------------------------
// Column l2norm of X[b][:,l] over DIM; write fp32 + bf16 copies.
// ---------------------------------------------------------------------------
__global__ __launch_bounds__(256) void norm_cols_kernel(
    const float* __restrict__ X, float* __restrict__ Xn, unsigned short* __restrict__ Xh)
{
    const int b = blockIdx.y;
    const int l = blockIdx.x * 256 + threadIdx.x;
    const float* xb = X + (size_t)b * DIM_ * L_ + l;
    float s = 0.f;
    for (int d = 0; d < DIM_; ++d) { float v = xb[(size_t)d * L_]; s += v * v; }
    const float inv = 1.f / fmaxf(sqrtf(s), 1e-12f);
    float* yb = Xn + (size_t)b * DIM_ * L_ + l;
    unsigned short* hb = Xh + (size_t)b * DIM_ * L_ + l;
    for (int d = 0; d < DIM_; ++d) {
        float v = xb[(size_t)d * L_] * inv;
        yb[(size_t)d * L_] = v;
        hb[(size_t)d * L_] = f2bf(v);
    }
}

__global__ void f2bf_kernel(const float* __restrict__ in, unsigned short* __restrict__ out, int n)
{
    int i = blockIdx.x * blockDim.x + threadIdx.x;
    if (i < n) out[i] = f2bf(in[i]);
}

// Row softmax over 1024 elements (one block per row), bf16 output.
__global__ __launch_bounds__(256) void softmax_rows_kernel(
    const float* __restrict__ S, unsigned short* __restrict__ P)
{
    __shared__ float red[256];
    const size_t row = blockIdx.x;
    const float* src = S + row * (size_t)L_;
    const int t = threadIdx.x;
    float x[4];
    float mx = -3.4e38f;
#pragma unroll
    for (int i = 0; i < 4; ++i) { x[i] = src[t + i * 256]; mx = fmaxf(mx, x[i]); }
    red[t] = mx; __syncthreads();
    for (int w = 128; w > 0; w >>= 1) { if (t < w) red[t] = fmaxf(red[t], red[t + w]); __syncthreads(); }
    mx = red[0]; __syncthreads();
    float sum = 0.f;
#pragma unroll
    for (int i = 0; i < 4; ++i) { x[i] = __expf(x[i] - mx); sum += x[i]; }
    red[t] = sum; __syncthreads();
    for (int w = 128; w > 0; w >>= 1) { if (t < w) red[t] += red[t + w]; __syncthreads(); }
    const float inv = 1.f / red[0];
    unsigned short* dst = P + row * (size_t)L_;
#pragma unroll
    for (int i = 0; i < 4; ++i) dst[t + i * 256] = f2bf(x[i] * inv);
}

// out = l2norm(l2norm(U1) + l2norm(U2)), all column-wise over DIM.
__global__ __launch_bounds__(256) void final_norm_kernel(
    const float* __restrict__ U1, const float* __restrict__ U2, float* __restrict__ out)
{
    const int b = blockIdx.y;
    const int l = blockIdx.x * 256 + threadIdx.x;
    const size_t base = (size_t)b * DIM_ * L_ + l;
    float s1 = 0.f, s2 = 0.f;
    for (int d = 0; d < DIM_; ++d) {
        float a = U1[base + (size_t)d * L_];
        float c = U2[base + (size_t)d * L_];
        s1 += a * a; s2 += c * c;
    }
    const float i1 = 1.f / fmaxf(sqrtf(s1), 1e-12f);
    const float i2 = 1.f / fmaxf(sqrtf(s2), 1e-12f);
    float sy = 0.f;
    for (int d = 0; d < DIM_; ++d) {
        float y = U1[base + (size_t)d * L_] * i1 + U2[base + (size_t)d * L_] * i2;
        sy += y * y;
    }
    const float iy = 1.f / fmaxf(sqrtf(sy), 1e-12f);
    for (int d = 0; d < DIM_; ++d) {
        float y = U1[base + (size_t)d * L_] * i1 + U2[base + (size_t)d * L_] * i2;
        out[base + (size_t)d * L_] = y * iy;
    }
}

// ---------------------------------------------------------------------------
extern "C" void kernel_launch(void* const* d_in, const int* in_sizes, int n_in,
                              void* d_out, int out_size, void* d_ws, size_t ws_size,
                              hipStream_t stream)
{
    (void)in_sizes; (void)n_in; (void)out_size; (void)ws_size;
    const float* V = (const float*)d_in[0];
    const float* T = (const float*)d_in[1];
    const float* Wm[6] = { (const float*)d_in[2], (const float*)d_in[3], (const float*)d_in[4],
                           (const float*)d_in[5], (const float*)d_in[6], (const float*)d_in[7] };

    const size_t BDL_F = (size_t)B_ * DIM_ * L_ * 4;
    const size_t BDL_H = BDL_F / 2;
    const size_t DD    = (size_t)DIM_ * DIM_;
    const size_t W_H   = 6 * DD * 2;
    const size_t BLL_F = (size_t)B_ * L_ * L_ * 4;
    const size_t BLL_H = BLL_F / 2;

    char* ws = (char*)d_ws;
    size_t o = 0;
    float*          Tn   = (float*)(ws + o);          o += BDL_F;
    unsigned short* Vbf  = (unsigned short*)(ws + o); o += BDL_H;
    unsigned short* Tnbf = (unsigned short*)(ws + o); o += BDL_H;
    unsigned short* Wbf  = (unsigned short*)(ws + o); o += W_H;
    unsigned short* Q1t  = (unsigned short*)(ws + o); o += BDL_H;   // [B][L][DIM]
    unsigned short* Q2t  = (unsigned short*)(ws + o); o += BDL_H;
    unsigned short* K1   = (unsigned short*)(ws + o); o += BDL_H;   // [B][DIM][L]
    unsigned short* V1x  = (unsigned short*)(ws + o); o += BDL_H;
    unsigned short* K2   = (unsigned short*)(ws + o); o += BDL_H;
    unsigned short* V2x  = (unsigned short*)(ws + o); o += BDL_H;
    float*          St   = (float*)(ws + o);          o += BLL_F;   // [B][L(k)][L(q)]
    unsigned short* Pt   = (unsigned short*)(ws + o); o += BLL_H;
    float*          U1   = (float*)(ws + o);          o += BDL_F;
    float*          U2   = (float*)(ws + o);          o += BDL_F;   // total ~583 MB

    const dim3 blk(256);
    const dim3 gridN(L_ / 256, B_);
    const dim3 gridG(L_ / 128, DIM_ / 128, B_);      // 8 x 8 x 16 workgroups

    // 1) Tn = l2norm(T) (+ bf16 copy); Vbf = bf16(V); Wbf = bf16(weights)
    norm_cols_kernel<<<gridN, blk, 0, stream>>>(T, Tn, Tnbf);
    const int nBDL = B_ * DIM_ * L_;
    f2bf_kernel<<<(nBDL + 255) / 256, 256, 0, stream>>>(V, Vbf, nBDL);
    for (int i = 0; i < 6; ++i)
        f2bf_kernel<<<((int)DD + 255) / 256, 256, 0, stream>>>(Wm[i], Wbf + i * DD, (int)DD);

    // 2) projections (relu fused); Q1/Q2 stored transposed for the QK GEMM
    gemm_proj_kernel<<<gridG, blk, 0, stream>>>(Wbf + 0 * DD, Vbf,  Q1t, 1);
    gemm_proj_kernel<<<gridG, blk, 0, stream>>>(Wbf + 1 * DD, Tnbf, K1,  0);
    gemm_proj_kernel<<<gridG, blk, 0, stream>>>(Wbf + 2 * DD, Tnbf, V1x, 0);
    gemm_proj_kernel<<<gridG, blk, 0, stream>>>(Wbf + 3 * DD, Tnbf, Q2t, 1);
    gemm_proj_kernel<<<gridG, blk, 0, stream>>>(Wbf + 4 * DD, Vbf,  K2,  0);
    gemm_proj_kernel<<<gridG, blk, 0, stream>>>(Wbf + 5 * DD, Vbf,  V2x, 0);

    // 3) branch 1: S1 = Q1^T K1 / 32 ; softmax over q ; U1 = V + V1 P1^T
    gemm_qk_kernel<<<gridG, blk, 0, stream>>>(Q1t, K1, St);
    softmax_rows_kernel<<<B_ * L_, 256, 0, stream>>>(St, Pt);
    gemm_pv_kernel<<<gridG, blk, 0, stream>>>(V1x, Pt, V, U1);

    // 4) branch 2: S2 = Q2^T K2 / 32 ; softmax over q ; U2 = Tn + V2 P2^T
    gemm_qk_kernel<<<gridG, blk, 0, stream>>>(Q2t, K2, St);
    softmax_rows_kernel<<<B_ * L_, 256, 0, stream>>>(St, Pt);
    gemm_pv_kernel<<<gridG, blk, 0, stream>>>(V2x, Pt, Tn, U2);

    // 5) out = l2norm(l2norm(U1) + l2norm(U2))
    final_norm_kernel<<<gridN, blk, 0, stream>>>(U1, U2, (float*)d_out);
}
